// DIAMSoftmaxLoss_13580686590550
// MI455X (gfx1250) — matrix operand order repaired and checked by reading, not compiled, not run on hardware
//
#include <hip/hip_runtime.h>
#include <hip/hip_bf16.h>

typedef __attribute__((ext_vector_type(16))) _Float16 v16h;
typedef __attribute__((ext_vector_type(8)))  float    v8f;

#define BS    1024
#define INF_  128
#define NC    100000
#define S_    30.0f
#define M_    0.35f
#define L2E   1.4426950408889634f
// arg = (S*logit - 30) * log2(e) = logit*(30*L2E) - 30*L2E
#define SL2E  43.2808512266689f

// ---------------- Kernel 1: normalize input rows, emit fp32 fn + packed f16 A-fragments
__global__ __launch_bounds__(128) void k_norm_in(const float* __restrict__ inp,
                                                 float* __restrict__ fn,
                                                 _Float16* __restrict__ fnA) {
    __shared__ float sw[4];
    int m = blockIdx.x, k = threadIdx.x;
    float x = inp[m * INF_ + k];
    float p = x * x;
    for (int o = 16; o >= 1; o >>= 1) p += __shfl_xor(p, o, 32);
    if ((k & 31) == 0) sw[k >> 5] = p;
    __syncthreads();
    float tot = sw[0] + sw[1] + sw[2] + sw[3];
    float d = fmaxf(sqrtf(tot), 1e-12f);
    float v = x / d;
    fn[m * INF_ + k] = v;
    // A fragment layout (16-bit A 16x32, wave32):
    // lane = M + 16*K[3];  half index h = K[2:0] + 8*K[4]
    int tile = m >> 4, mm = m & 15, kc = k >> 5, kk = k & 31;
    int lane = mm + (((kk >> 3) & 1) << 4);
    int h    = (kk & 7) + ((kk >> 4) << 3);
    fnA[((((tile << 2) + kc) << 5) + lane) * 16 + h] = (_Float16)v;
}

// ---------------- Kernel 2: per-class reciprocal L2 norm (one wave per row)
__global__ __launch_bounds__(256) void k_rnorm(const float* __restrict__ weight,
                                               float* __restrict__ rnorm) {
    int row  = blockIdx.x * 8 + (threadIdx.x >> 5);
    int lane = threadIdx.x & 31;
    const float4* w4 = (const float4*)weight;
    float4 q = w4[row * 32 + lane];
    float p = q.x * q.x + q.y * q.y + q.z * q.z + q.w * q.w;
    for (int o = 16; o >= 1; o >>= 1) p += __shfl_xor(p, o, 32);
    if (lane == 0) rnorm[row] = 1.0f / fmaxf(sqrtf(p), 1e-12f);
}

// ---------------- Kernel 3: zero sumExp + counts
__global__ __launch_bounds__(256) void k_zero(float* __restrict__ sumExp, int* __restrict__ counts) {
    int i = blockIdx.x * 256 + threadIdx.x;
    if (i < BS) sumExp[i] = 0.0f;
    if (i < NC) counts[i] = 0;
}

// ---------------- Kernel 4: exact fp32 positive logit per row (one wave per row)
__global__ __launch_bounds__(256) void k_pos(const float* __restrict__ fn,
                                             const int* __restrict__ label,
                                             const float* __restrict__ weight,
                                             const float* __restrict__ rnorm,
                                             float* __restrict__ pos) {
    int b    = blockIdx.x * 8 + (threadIdx.x >> 5);
    int lane = threadIdx.x & 31;
    int l = label[b];
    const float4* f4 = (const float4*)fn;
    const float4* w4 = (const float4*)weight;
    float4 f = f4[b * 32 + lane];
    float4 w = w4[l * 32 + lane];
    float p = f.x * w.x + f.y * w.y + f.z * w.z + f.w * w.w;
    for (int o = 16; o >= 1; o >>= 1) p += __shfl_xor(p, o, 32);
    if (lane == 0) pos[b] = S_ * p * rnorm[l];
}

// ---------------- Kernel 5: fused GEMM + exp + per-row partial sums (WMMA f16)
// 128 classes per workgroup, all 1024 batch rows; 256 threads = 8 waves.
__global__ __launch_bounds__(256) void k_gemm(const float* __restrict__ weight,
                                              const float* __restrict__ rnorm,
                                              const v16h* __restrict__ fnA,
                                              float* __restrict__ sumExp) {
    // B fragments for 8 class-tiles x 4 K-chunks: 8*4*32 v16h = 32 KB
    __shared__ v16h Bf[8 * 4 * 32];
    const int c0  = blockIdx.x * 128;
    const int tid = threadIdx.x;

    // Fill LDS: normalize + convert 128 classes (pad with 0 past NC)
    _Float16* Bh = (_Float16*)Bf;
    for (int i = tid; i < 128 * 128; i += 256) {
        int cl = i >> 7, k = i & 127;
        int c = c0 + cl;
        float v = 0.0f;
        if (c < NC) v = weight[c * INF_ + k] * rnorm[c];
        // B fragment layout (16-bit B 32x16): lane = N + 16*K[3]; half = K[2:0] + 8*K[4]
        int ct = cl >> 4, n = cl & 15;
        int kc = k >> 5, kk = k & 31;
        int lane = n + (((kk >> 3) & 1) << 4);
        int h    = (kk & 7) + ((kk >> 4) << 3);
        Bh[((((ct << 2) + kc) << 5) + lane) * 16 + h] = (_Float16)v;
    }
    __syncthreads();

    const int wave = tid >> 5, lane = tid & 31;
    for (int mt = wave; mt < BS / 16; mt += 8) {
        // A fragments for this 16-row batch tile (K = 0..127)
        v16h a0 = fnA[((mt << 2) + 0) * 32 + lane];
        v16h a1 = fnA[((mt << 2) + 1) * 32 + lane];
        v16h a2 = fnA[((mt << 2) + 2) * 32 + lane];
        v16h a3 = fnA[((mt << 2) + 3) * 32 + lane];
        float acc[8];
#pragma unroll
        for (int r = 0; r < 8; ++r) acc[r] = 0.0f;

        for (int ct = 0; ct < 8; ++ct) {
            v8f cc = {};
            cc = __builtin_amdgcn_wmma_f32_16x16x32_f16(false, a0, false,
                     Bf[((ct << 2) + 0) * 32 + lane], (short)0, cc, false, false);
            cc = __builtin_amdgcn_wmma_f32_16x16x32_f16(false, a1, false,
                     Bf[((ct << 2) + 1) * 32 + lane], (short)0, cc, false, false);
            cc = __builtin_amdgcn_wmma_f32_16x16x32_f16(false, a2, false,
                     Bf[((ct << 2) + 2) * 32 + lane], (short)0, cc, false, false);
            cc = __builtin_amdgcn_wmma_f32_16x16x32_f16(false, a3, false,
                     Bf[((ct << 2) + 3) * 32 + lane], (short)0, cc, false, false);
            // this lane's class for the whole tile
            int cg = c0 + (ct << 4) + (lane & 15);
            float vm = (cg < NC) ? 1.0f : 0.0f;
#pragma unroll
            for (int r = 0; r < 8; ++r) {
                float arg = fmaf(cc[r], SL2E, -SL2E);   // (S*logit - 30)*log2e
                acc[r] += vm * __builtin_amdgcn_exp2f(arg);
            }
        }
        // Reduce over the 16 classes held by each half-wave; rows:
        // reg r, lanes 0-15 -> M = r ; lanes 16-31 -> M = r + 8
#pragma unroll
        for (int r = 0; r < 8; ++r) {
            float s = acc[r];
            s += __shfl_xor(s, 1, 32);
            s += __shfl_xor(s, 2, 32);
            s += __shfl_xor(s, 4, 32);
            s += __shfl_xor(s, 8, 32);
            if ((lane & 15) == 0) {
                int row = (mt << 4) + r + ((lane >> 4) << 3);
                atomicAdd(&sumExp[row], s);
            }
        }
    }
}

// ---------------- Kernel 6: loss = mean softplus(M + lse_neg - pos)
__global__ __launch_bounds__(1024) void k_loss(const float* __restrict__ sumExp,
                                               const float* __restrict__ pos,
                                               float* __restrict__ out) {
    __shared__ float sd[1024];
    int t = threadIdx.x;
    float sE = sumExp[t], p = pos[t];
    float ep   = __builtin_amdgcn_exp2f((p - 30.0f) * L2E); // e^(pos-30)
    float diff = fmaxf(sE - ep, 1e-38f);
    float lse  = 30.0f + logf(diff);
    float z    = M_ + lse - p;
    sd[t] = (z > 20.0f) ? z : log1pf(expf(z));
    __syncthreads();
    for (int s = 512; s > 0; s >>= 1) {
        if (t < s) sd[t] += sd[t + s];
        __syncthreads();
    }
    if (t == 0) out[0] = sd[0] * (1.0f / (float)BS);
}

// ---------------- Weight-update kernels (accumulate directly into d_out+1)
__global__ __launch_bounds__(256) void k_count(const int* __restrict__ label,
                                               int* __restrict__ counts) {
    int b = blockIdx.x * 256 + threadIdx.x;
    if (b < BS) atomicAdd(&counts[label[b]], 1);
}

__global__ __launch_bounds__(256) void k_zero_present(const int* __restrict__ label,
                                                      float* __restrict__ Wout) {
    int i = blockIdx.x * 256 + threadIdx.x;   // i < BS*INF_
    int b = i >> 7, f = i & 127;
    Wout[label[b] * INF_ + f] = 0.0f;         // races all write 0 -> benign
}

__global__ __launch_bounds__(256) void k_scatter(const int* __restrict__ label,
                                                 const float* __restrict__ fn,
                                                 float* __restrict__ Wout) {
    int i = blockIdx.x * 256 + threadIdx.x;
    int b = i >> 7, f = i & 127;
    atomicAdd(&Wout[label[b] * INF_ + f], fn[i]);
}

__global__ __launch_bounds__(256) void k_final(const int* __restrict__ counts,
                                               const float* __restrict__ weight,
                                               float* __restrict__ Wout) {
    int i = blockIdx.x * 256 + threadIdx.x;   // i < NC*INF_
    if (i >= NC * INF_) return;
    int c = i >> 7;
    int cnt = counts[c];
    Wout[i] = (cnt > 0) ? (Wout[i] / (float)cnt) : weight[i];
}

extern "C" void kernel_launch(void* const* d_in, const int* in_sizes, int n_in,
                              void* d_out, int out_size, void* d_ws, size_t ws_size,
                              hipStream_t stream) {
    const float* inp    = (const float*)d_in[0];
    const int*   label  = (const int*)d_in[1];
    const float* weight = (const float*)d_in[2];

    float* out  = (float*)d_out;       // [0] = loss
    float* Wout = out + 1;             // [1..] = new_weight (NC*INF_ floats)

    // workspace layout (~1.6 MB)
    char* w = (char*)d_ws;
    float*    fn     = (float*)(w);                 // 1024*128*4 = 524288
    _Float16* fnA    = (_Float16*)(w + 524288);     // 1024*128*2 = 262144
    float*    rnorm  = (float*)(w + 786432);        // 100000*4   = 400000
    float*    sumExp = (float*)(w + 1187840);       // 1024*4
    float*    pos    = (float*)(w + 1191936);       // 1024*4
    int*      counts = (int*)(w + 1196032);         // 100000*4

    k_norm_in<<<BS, 128, 0, stream>>>(inp, fn, fnA);
    k_rnorm<<<NC / 8, 256, 0, stream>>>(weight, rnorm);
    k_zero<<<(NC + 255) / 256, 256, 0, stream>>>(sumExp, counts);
    k_pos<<<BS / 8, 256, 0, stream>>>(fn, label, weight, rnorm, pos);
    k_gemm<<<(NC + 127) / 128, 256, 0, stream>>>(weight, rnorm, (const v16h*)fnA, sumExp);
    k_loss<<<1, 1024, 0, stream>>>(sumExp, pos, out);

    k_count<<<BS / 256, 256, 0, stream>>>(label, counts);
    k_zero_present<<<(BS * INF_) / 256, 256, 0, stream>>>(label, Wout);
    k_scatter<<<(BS * INF_) / 256, 256, 0, stream>>>(label, fn, Wout);
    k_final<<<(NC * INF_ + 255) / 256, 256, 0, stream>>>(counts, weight, Wout);
}
